// CrossAttention_26250840113414
// MI455X (gfx1250) — matrix operand order repaired and checked by reading, not compiled, not run on hardware
//
#include <hip/hip_runtime.h>
#include <hip/hip_bf16.h>

// Problem dimensions (fixed by the reference harness)
constexpr int BB = 16;    // batch
constexpr int DD = 1024;  // document length
constexpr int QQ = 128;   // query length
constexpr int HH = 1024;  // hidden
constexpr float NEG_INF = -1e30f;

typedef __attribute__((ext_vector_type(8)))  float          v8f;
typedef __attribute__((ext_vector_type(8)))  unsigned short v8u;
typedef __attribute__((ext_vector_type(16))) unsigned short v16u;
typedef __attribute__((ext_vector_type(16))) __bf16         v16bf;

__device__ __forceinline__ unsigned short f32_to_bf16(float f) {
    unsigned int u = __float_as_uint(f);
    unsigned int r = u + 0x7FFFu + ((u >> 16) & 1u);   // round-to-nearest-even
    return (unsigned short)(r >> 16);
}

__device__ __forceinline__ v16u cat16(v8u lo, v8u hi) {
    return __builtin_shufflevector(lo, hi, 0,1,2,3,4,5,6,7,8,9,10,11,12,13,14,15);
}

__device__ __forceinline__ v8f wmma_bf16(v16u a, v16u b, v8f c) {
    return __builtin_amdgcn_wmma_f32_16x16x32_bf16(
        /*neg_a=*/false, __builtin_bit_cast(v16bf, a),
        /*neg_b=*/false, __builtin_bit_cast(v16bf, b),
        /*c_mod=*/(short)0, c, /*reuse_a=*/false, /*reuse_b=*/false);
}

// ---------------------------------------------------------------------------
// 1a) Per-(b,d) row: s_d = U_d . w_d  and Udw_bf16 = bf16(U_d * w_dot)
// ---------------------------------------------------------------------------
__global__ void prep_d_kernel(const float* __restrict__ ud, const float* __restrict__ wcw,
                              unsigned short* __restrict__ udw, float* __restrict__ sd) {
    const int d = blockIdx.x, b = blockIdx.y, t = threadIdx.x;  // 256 threads
    const float* row = ud + ((size_t)b * DD + d) * HH;
    unsigned short* orow = udw + ((size_t)b * DD + d) * HH;
    float acc = 0.f;
    #pragma unroll
    for (int i = 0; i < HH; i += 256) {
        int h = i + t;
        float u = row[h];
        acc += u * wcw[h];                             // w_d
        orow[h] = f32_to_bf16(u * wcw[2 * HH + h]);    // w_dot
    }
    __shared__ float red[256];
    red[t] = acc; __syncthreads();
    for (int o = 128; o > 0; o >>= 1) { if (t < o) red[t] += red[t + o]; __syncthreads(); }
    if (t == 0) sd[b * DD + d] = red[0];
}

// ---------------------------------------------------------------------------
// 1b) Per-(b,q) row: s_q = U_q . w_q  and Uq_bf16 = bf16(U_q)
// ---------------------------------------------------------------------------
__global__ void prep_q_kernel(const float* __restrict__ uq, const float* __restrict__ wcw,
                              unsigned short* __restrict__ uq16, float* __restrict__ sq) {
    const int q = blockIdx.x, b = blockIdx.y, t = threadIdx.x;  // 256 threads
    const float* row = uq + ((size_t)b * QQ + q) * HH;
    unsigned short* orow = uq16 + ((size_t)b * QQ + q) * HH;
    float acc = 0.f;
    #pragma unroll
    for (int i = 0; i < HH; i += 256) {
        int h = i + t;
        float u = row[h];
        acc += u * wcw[HH + h];                        // w_q
        orow[h] = f32_to_bf16(u);
    }
    __shared__ float red[256];
    red[t] = acc; __syncthreads();
    for (int o = 128; o > 0; o >>= 1) { if (t < o) red[t] += red[t + o]; __syncthreads(); }
    if (t == 0) sq[b * QQ + q] = red[0];
}

// ---------------------------------------------------------------------------
// 2) LDS-tiled transpose: src f32 (B, R, C) -> dst bf16 (B, C, R)
// ---------------------------------------------------------------------------
__global__ void transpose_f32_bf16(const float* __restrict__ src, unsigned short* __restrict__ dst,
                                   int R, int C) {
    __shared__ float tile[32][33];
    const int b = blockIdx.z;
    const int c0 = blockIdx.x * 32, r0 = blockIdx.y * 32;
    const int tx = threadIdx.x, ty = threadIdx.y;               // (32, 8)
    const float* s = src + (size_t)b * R * C;
    unsigned short* dp = dst + (size_t)b * R * C;
    #pragma unroll
    for (int j = 0; j < 32; j += 8)
        tile[ty + j][tx] = s[(size_t)(r0 + ty + j) * C + c0 + tx];
    __syncthreads();
    #pragma unroll
    for (int j = 0; j < 32; j += 8)
        dp[(size_t)(c0 + ty + j) * R + r0 + tx] = f32_to_bf16(tile[tx][ty + j]);
}

// ---------------------------------------------------------------------------
// 3) Score GEMM: S[b,d,q] = (U_d*w_dot) . U_q  + s_d[d] + s_q[q] + bias
//    A = Udw_bf16 (D,H) row-major, B = Uq_bf16^T (columns are U_q rows: contig in h)
//    One wave per 16x16 tile, K-loop over H in steps of 32.
// ---------------------------------------------------------------------------
__global__ void score_kernel(const unsigned short* __restrict__ udw,
                             const unsigned short* __restrict__ uq16,
                             const float* __restrict__ sd, const float* __restrict__ sq,
                             const float* __restrict__ wcb, float* __restrict__ S) {
    const int lane = threadIdx.x;                 // block = 32 (one wave)
    const int q0 = blockIdx.x * 16, d0 = blockIdx.y * 16, b = blockIdx.z;
    const int half = lane >> 4, l16 = lane & 15;
    const unsigned short* arow = udw  + ((size_t)b * DD + (d0 + l16)) * HH;
    const unsigned short* brow = uq16 + ((size_t)b * QQ + (q0 + l16)) * HH;
    const int akoff = half * 16;      // A lanes>=16 carry K+16
    const int bkoff = half * 16;      // B lanes>=16 carry K+16

    v8f c = {};
    for (int k = 0; k < HH; k += 32) {
        v8u alo = *(const v8u*)(arow + k + akoff);
        v8u ahi = *(const v8u*)(arow + k + akoff + 8);
        v8u blo = *(const v8u*)(brow + k + bkoff);
        v8u bhi = *(const v8u*)(brow + k + bkoff + 8);
        c = wmma_bf16(cat16(alo, ahi), cat16(blo, bhi), c);
    }
    const float bias = wcb[0];
    const float sqv  = sq[b * QQ + q0 + l16];
    #pragma unroll
    for (int r = 0; r < 8; ++r) {
        const int m = r + half * 8;
        const int row = d0 + m;
        S[((size_t)b * DD + row) * QQ + q0 + l16] = c[r] + sd[b * DD + row] + sqv + bias;
    }
}

// ---------------------------------------------------------------------------
// 4a) Masked softmax over q (axis -1). One 128-thread block per (b,d) row.
//     Output: Sd2q bf16 (D,Q) row-major.
// ---------------------------------------------------------------------------
__global__ void softmax_d2q_kernel(const float* __restrict__ S,
                                   const int* __restrict__ qmask, const int* __restrict__ dmask,
                                   unsigned short* __restrict__ out) {
    const int d = blockIdx.x, b = blockIdx.y, t = threadIdx.x;  // 128 threads
    __shared__ float red[128];
    const bool on = (dmask[b * DD + d] > 0) && (qmask[b * QQ + t] > 0);
    const float s = S[((size_t)b * DD + d) * QQ + t];
    const float logit = on ? s : NEG_INF;
    red[t] = logit; __syncthreads();
    for (int o = 64; o > 0; o >>= 1) { if (t < o) red[t] = fmaxf(red[t], red[t + o]); __syncthreads(); }
    const float mx = red[0]; __syncthreads();
    const float e = __expf(logit - mx);
    red[t] = e; __syncthreads();
    for (int o = 64; o > 0; o >>= 1) { if (t < o) red[t] += red[t + o]; __syncthreads(); }
    const float p = on ? e / red[0] : 0.f;
    out[((size_t)b * DD + d) * QQ + t] = f32_to_bf16(p);
}

// ---------------------------------------------------------------------------
// 4b) Masked softmax over d (axis -2). One 256-thread block per (b,q) column.
//     Output TRANSPOSED: Sq2dT bf16 (Q,D) row-major (A-fragment-ready).
// ---------------------------------------------------------------------------
__global__ void softmax_q2d_kernel(const float* __restrict__ S,
                                   const int* __restrict__ qmask, const int* __restrict__ dmask,
                                   unsigned short* __restrict__ outT) {
    const int q = blockIdx.x, b = blockIdx.y, t = threadIdx.x;  // 256 threads, 4 d each
    __shared__ float red[256];
    const bool qm = qmask[b * QQ + q] > 0;
    float logit[4]; bool on[4];
    float mx = NEG_INF;
    #pragma unroll
    for (int i = 0; i < 4; ++i) {
        const int d = t + i * 256;
        on[i] = qm && (dmask[b * DD + d] > 0);
        const float s = S[((size_t)b * DD + d) * QQ + q];
        logit[i] = on[i] ? s : NEG_INF;
        mx = fmaxf(mx, logit[i]);
    }
    red[t] = mx; __syncthreads();
    for (int o = 128; o > 0; o >>= 1) { if (t < o) red[t] = fmaxf(red[t], red[t + o]); __syncthreads(); }
    mx = red[0]; __syncthreads();
    float e[4], sum = 0.f;
    #pragma unroll
    for (int i = 0; i < 4; ++i) { e[i] = __expf(logit[i] - mx); sum += e[i]; }
    red[t] = sum; __syncthreads();
    for (int o = 128; o > 0; o >>= 1) { if (t < o) red[t] += red[t + o]; __syncthreads(); }
    sum = red[0];
    #pragma unroll
    for (int i = 0; i < 4; ++i) {
        const int d = t + i * 256;
        const float p = on[i] ? e[i] / sum : 0.f;
        outT[((size_t)b * QQ + q) * DD + d] = f32_to_bf16(p);
    }
}

// ---------------------------------------------------------------------------
// 5) W GEMM: W = S_q2d^T (Q,D) @ U_d (D,H); A = Sq2dT rows, B cols = UdT rows.
//    Output written TRANSPOSED as Wt bf16 (H,Q) so it is B-ready for A_q2d.
// ---------------------------------------------------------------------------
__global__ void gemm_w_kernel(const unsigned short* __restrict__ sq2dT,
                              const unsigned short* __restrict__ udT,
                              unsigned short* __restrict__ wt) {
    const int lane = threadIdx.x;
    const int q0 = blockIdx.x * 16, h0 = blockIdx.y * 16, b = blockIdx.z;
    const int half = lane >> 4, l16 = lane & 15;
    const unsigned short* arow = sq2dT + ((size_t)b * QQ + q0 + l16) * DD;
    const unsigned short* brow = udT   + ((size_t)b * HH + h0 + l16) * DD;
    const int akoff = half * 16, bkoff = half * 16;

    v8f c = {};
    for (int k = 0; k < DD; k += 32) {
        v8u alo = *(const v8u*)(arow + k + akoff);
        v8u ahi = *(const v8u*)(arow + k + akoff + 8);
        v8u blo = *(const v8u*)(brow + k + bkoff);
        v8u bhi = *(const v8u*)(brow + k + bkoff + 8);
        c = wmma_bf16(cat16(alo, ahi), cat16(blo, bhi), c);
    }
    // Element (M=r+8*half, N=l16) -> Wt[h0+N][q0+M]: 8 contiguous bf16 per lane
    v8u pk;
    #pragma unroll
    for (int r = 0; r < 8; ++r) pk[r] = f32_to_bf16(c[r]);
    *(v8u*)(wt + ((size_t)b * HH + h0 + l16) * QQ + q0 + half * 8) = pk;
}

// ---------------------------------------------------------------------------
// 6) A_d2q GEMM: A = Sd2q (D,Q) @ U_q (Q,H); B cols = UqT rows. K = 128.
//    Fused epilogue: V[:,0:H]=U_d, V[:,H:2H]=A, V[:,2H:3H]=U_d*A
// ---------------------------------------------------------------------------
__global__ void gemm_ad2q_kernel(const unsigned short* __restrict__ sd2q,
                                 const unsigned short* __restrict__ uqT,
                                 const float* __restrict__ ud, float* __restrict__ V) {
    const int lane = threadIdx.x;
    const int h0 = blockIdx.x * 16, d0 = blockIdx.y * 16, b = blockIdx.z;
    const int half = lane >> 4, l16 = lane & 15;
    const unsigned short* arow = sd2q + ((size_t)b * DD + d0 + l16) * QQ;
    const unsigned short* brow = uqT  + ((size_t)b * HH + h0 + l16) * QQ;
    const int akoff = half * 16, bkoff = half * 16;

    v8f c = {};
    #pragma unroll
    for (int k = 0; k < QQ; k += 32) {
        v8u alo = *(const v8u*)(arow + k + akoff);
        v8u ahi = *(const v8u*)(arow + k + akoff + 8);
        v8u blo = *(const v8u*)(brow + k + bkoff);
        v8u bhi = *(const v8u*)(brow + k + bkoff + 8);
        c = wmma_bf16(cat16(alo, ahi), cat16(blo, bhi), c);
    }
    const float* udb = ud + (size_t)b * DD * HH;
    float* Vb = V + (size_t)b * DD * (4 * HH);
    const int hh = h0 + l16;
    #pragma unroll
    for (int r = 0; r < 8; ++r) {
        const int dd = d0 + r + half * 8;
        const float u = udb[(size_t)dd * HH + hh];
        const float a = c[r];
        float* vrow = Vb + (size_t)dd * (4 * HH);
        vrow[hh]          = u;
        vrow[HH + hh]     = a;
        vrow[2 * HH + hh] = u * a;
    }
}

// ---------------------------------------------------------------------------
// 7) A_q2d GEMM: A = Sd2q (D,Q) @ W (Q,H); B cols = Wt rows. K = 128.
//    Fused epilogue: V[:,3H:4H] = U_d * A_q2d
// ---------------------------------------------------------------------------
__global__ void gemm_aq2d_kernel(const unsigned short* __restrict__ sd2q,
                                 const unsigned short* __restrict__ wt,
                                 const float* __restrict__ ud, float* __restrict__ V) {
    const int lane = threadIdx.x;
    const int h0 = blockIdx.x * 16, d0 = blockIdx.y * 16, b = blockIdx.z;
    const int half = lane >> 4, l16 = lane & 15;
    const unsigned short* arow = sd2q + ((size_t)b * DD + d0 + l16) * QQ;
    const unsigned short* brow = wt   + ((size_t)b * HH + h0 + l16) * QQ;
    const int akoff = half * 16, bkoff = half * 16;

    v8f c = {};
    #pragma unroll
    for (int k = 0; k < QQ; k += 32) {
        v8u alo = *(const v8u*)(arow + k + akoff);
        v8u ahi = *(const v8u*)(arow + k + akoff + 8);
        v8u blo = *(const v8u*)(brow + k + bkoff);
        v8u bhi = *(const v8u*)(brow + k + bkoff + 8);
        c = wmma_bf16(cat16(alo, ahi), cat16(blo, bhi), c);
    }
    const float* udb = ud + (size_t)b * DD * HH;
    float* Vb = V + (size_t)b * DD * (4 * HH);
    const int hh = h0 + l16;
    #pragma unroll
    for (int r = 0; r < 8; ++r) {
        const int dd = d0 + r + half * 8;
        const float u = udb[(size_t)dd * HH + hh];
        Vb[(size_t)dd * (4 * HH) + 3 * HH + hh] = u * c[r];
    }
}

// ---------------------------------------------------------------------------
extern "C" void kernel_launch(void* const* d_in, const int* in_sizes, int n_in,
                              void* d_out, int out_size, void* d_ws, size_t ws_size,
                              hipStream_t stream) {
    (void)in_sizes; (void)n_in; (void)out_size; (void)ws_size;
    const float* U_d   = (const float*)d_in[0];
    const float* U_q   = (const float*)d_in[1];
    const float* wc_w  = (const float*)d_in[2];
    const float* wc_b  = (const float*)d_in[3];
    const int*   qmask = (const int*)d_in[4];
    const int*   dmask = (const int*)d_in[5];
    float* V = (float*)d_out;

    // Workspace layout (bytes); total ~92.2 MB
    char* ws = (char*)d_ws;
    size_t off = 0;
    unsigned short* udw   = (unsigned short*)(ws + off); off += (size_t)BB * DD * HH * 2; // 32 MB
    unsigned short* uq16  = (unsigned short*)(ws + off); off += (size_t)BB * QQ * HH * 2; //  4 MB
    unsigned short* udT   = (unsigned short*)(ws + off); off += (size_t)BB * HH * DD * 2; // 32 MB
    unsigned short* uqT   = (unsigned short*)(ws + off); off += (size_t)BB * HH * QQ * 2; //  4 MB
    float*          sd    = (float*)(ws + off);          off += (size_t)BB * DD * 4;      // 64 KB
    float*          sq    = (float*)(ws + off);          off += (size_t)BB * QQ * 4;      //  8 KB
    float*          S     = (float*)(ws + off);          off += (size_t)BB * DD * QQ * 4; //  8 MB
    unsigned short* sd2q  = (unsigned short*)(ws + off); off += (size_t)BB * DD * QQ * 2; //  4 MB
    unsigned short* sq2dT = (unsigned short*)(ws + off); off += (size_t)BB * QQ * DD * 2; //  4 MB
    unsigned short* wt    = (unsigned short*)(ws + off); off += (size_t)BB * HH * QQ * 2; //  4 MB

    // 1) row scalars + bf16 conversions
    prep_d_kernel<<<dim3(DD, BB), 256, 0, stream>>>(U_d, wc_w, udw, sd);
    prep_q_kernel<<<dim3(QQ, BB), 256, 0, stream>>>(U_q, wc_w, uq16, sq);

    // 2) transposes: U_d (D,H) -> udT (H,D);  U_q (Q,H) -> uqT (H,Q)
    transpose_f32_bf16<<<dim3(HH / 32, DD / 32, BB), dim3(32, 8), 0, stream>>>(U_d, udT, DD, HH);
    transpose_f32_bf16<<<dim3(HH / 32, QQ / 32, BB), dim3(32, 8), 0, stream>>>(U_q, uqT, QQ, HH);

    // 3) similarity scores S (WMMA, K=H=1024)
    score_kernel<<<dim3(QQ / 16, DD / 16, BB), 32, 0, stream>>>(udw, uq16, sd, sq, wc_b, S);

    // 4) masked softmaxes
    softmax_d2q_kernel<<<dim3(DD, BB), 128, 0, stream>>>(S, qmask, dmask, sd2q);
    softmax_q2d_kernel<<<dim3(QQ, BB), 256, 0, stream>>>(S, qmask, dmask, sq2dT);

    // 5) W^T = (S_q2d^T @ U_d)^T  (WMMA, K=D=1024)
    gemm_w_kernel<<<dim3(QQ / 16, HH / 16, BB), 32, 0, stream>>>(sq2dT, udT, wt);

    // 6) A_d2q + fused concat writes (WMMA, K=Q=128)
    gemm_ad2q_kernel<<<dim3(HH / 16, DD / 16, BB), 32, 0, stream>>>(sd2q, uqT, U_d, V);

    // 7) A_q2d + fused concat write (WMMA, K=Q=128)
    gemm_aq2d_kernel<<<dim3(HH / 16, DD / 16, BB), 32, 0, stream>>>(sd2q, wt, U_d, V);
}
